// MaskGRU_51333449122320
// MI455X (gfx1250) — matrix operand order repaired and checked by reading, not compiled, not run on hardware
//
#include <hip/hip_runtime.h>
#include <hip/hip_bf16.h>

#define T_DIM 512
#define B_DIM 64
#define I_DIM 512
#define H_DIM 512
#define G3    (3 * H_DIM)     // 1536 gate rows
#define NWG   16              // scan workgroups
#define HC    (H_DIM / NWG)   // 32 hidden columns per WG
#define BSTR  68              // padded batch stride in gh LDS (bank-conflict avoidance)

typedef __attribute__((ext_vector_type(16))) __bf16 v16bf;
typedef __attribute__((ext_vector_type(8)))  float  v8f;
typedef __attribute__((ext_vector_type(4)))  unsigned int u32x4;
typedef __attribute__((ext_vector_type(8)))  int i32x8;
typedef __attribute__((ext_vector_type(4)))  int i32x4;

union FragU {
    struct { uint4 lo, hi; } u;
    v16bf v;
};

__device__ __forceinline__ v8f zero8() {
    v8f z = {0.f, 0.f, 0.f, 0.f, 0.f, 0.f, 0.f, 0.f};
    return z;
}

__device__ __forceinline__ v8f wmma_bf16(v16bf a, v16bf b, v8f c) {
    return __builtin_amdgcn_wmma_f32_16x16x32_bf16(false, a, false, b, (short)0, c,
                                                   false, false);
}

// A-matrix fragment (16x32 bf16, M x K), row-major source with leading dim ld.
// Lane l (l<16): row = l, K = {0..7, 16..23}; lane l+16: row = l, K = {8..15, 24..31}.
__device__ __forceinline__ v16bf load_frag_a(const __bf16* __restrict__ A, int ld,
                                             int row, int k0, int lane) {
    const int q = (lane < 16) ? 0 : 8;
    const __bf16* p = A + (size_t)row * ld + k0 + q;
    FragU f;
    f.u.lo = *reinterpret_cast<const uint4*>(p);
    f.u.hi = *reinterpret_cast<const uint4*>(p + 16);
    return f.v;
}

// Same fragment shape, but source lives in LDS (lowered to ds_load_b128).
__device__ __forceinline__ v16bf load_frag_a_lds(const __bf16* A, int ld,
                                                 int row, int k0, int lane) {
    const int q = (lane < 16) ? 0 : 8;
    const __bf16* p = A + (size_t)row * ld + k0 + q;
    FragU f;
    f.u.lo = *reinterpret_cast<const uint4*>(p);
    f.u.hi = *reinterpret_cast<const uint4*>(p + 16);
    return f.v;
}

// B-matrix fragment (32x16 bf16, K x N) where B[k][n] = W[n][k] (W row-major, ld = K).
// Lane l: column n = l%16, K = (l<16 ? {0..7, 16..23} : {8..15, 24..31}).
__device__ __forceinline__ v16bf load_frag_b(const __bf16* __restrict__ W, int ld,
                                             int nrow, int k0, int lane) {
    const int q = (lane < 16) ? 0 : 8;
    const __bf16* p = W + (size_t)nrow * ld + k0 + q;
    FragU f;
    f.u.lo = *reinterpret_cast<const uint4*>(p);
    f.u.hi = *reinterpret_cast<const uint4*>(p + 16);
    return f.v;
}

// 1D TDM copy: nelem2B bf16 elements from global gsrc -> LDS ldst.
// D# built per CDNA5 ISA ch.8: group0 {count, lds_addr, global_addr, type=2},
// group1 {data_size=2B, tensor_dim0=tile_dim0=n, stride0=n}, groups 2/3 zero
// (1D tile). 6-arg builtin form (clang-23 / therock-10.0 headers).
__device__ __forceinline__ void tdm_load_1d(void* ldst, const void* gsrc,
                                            unsigned nelem2B) {
    const unsigned lds_base = (unsigned)(size_t)ldst;
    const unsigned long long ga = (unsigned long long)(size_t)gsrc;
    u32x4 g0;
    g0[0] = 1u;                                              // count=1 (valid), no gather
    g0[1] = lds_base;                                        // lds_addr
    g0[2] = (unsigned)(ga & 0xFFFFFFFFu);                    // global_addr[31:0]
    g0[3] = (unsigned)((ga >> 32) & 0x1FFFFFFu) | (2u << 30); // addr[56:32] | type=2
    i32x8 g1;
    g1[0] = (int)(1u << 16);                                 // data_size=1 -> 2 bytes
    g1[1] = (int)((nelem2B & 0xFFFFu) << 16);                // tensor_dim0[15:0]
    g1[2] = (int)(((nelem2B >> 16) & 0xFFFFu) | (1u << 16)); // dim0[31:16], tensor_dim1=1
    g1[3] = (int)((nelem2B & 0xFFFFu) << 16);                // tile_dim0 = nelem2B
    g1[4] = 0;                                               // tile_dim1/2 unused
    g1[5] = (int)nelem2B;                                    // tensor_dim0_stride[31:0]
    g1[6] = 0;
    g1[7] = 0;
    i32x4 gz4 = {0, 0, 0, 0};
    i32x8 gz8 = {0, 0, 0, 0, 0, 0, 0, 0};
    __builtin_amdgcn_tensor_load_to_lds(g0, g1, gz4, gz4, gz8, 0);
}

__global__ void init_counter_kernel(unsigned* __restrict__ c) {
    if (threadIdx.x == 0 && blockIdx.x == 0) *c = 0u;
}

__global__ void convert_bf16_kernel(const float* __restrict__ in,
                                    __bf16* __restrict__ out, int n) {
    int i = (blockIdx.x * blockDim.x + threadIdx.x) * 4;
    if (i + 3 < n) {
        float4 v = *reinterpret_cast<const float4*>(in + i);
        union { __bf16 h[4]; uint2 u; } o;
        o.h[0] = (__bf16)v.x;
        o.h[1] = (__bf16)v.y;
        o.h[2] = (__bf16)v.z;
        o.h[3] = (__bf16)v.w;
        *reinterpret_cast<uint2*>(out + i) = o.u;
    }
}

// gx[m][n] = sum_k A[m][k] * W[n][k] + bias[n]
// M = 32768, N = 1536, K = 512. Each wave: 64x32 output tile (4 m-tiles x
// 2 n-tiles). Two-stage ping-pong K pipeline (k += 64), no register rotation:
// loads for stage s+1 are issued before the WMMAs consuming stage s.
__global__ void __launch_bounds__(256) gx_gemm_kernel(
    const __bf16* __restrict__ A, const __bf16* __restrict__ W,
    const float* __restrict__ bias, float* __restrict__ gx) {
    const int lane = threadIdx.x & 31;
    const int wave = threadIdx.x >> 5;
    const int job  = blockIdx.x * 8 + wave;   // 24576 jobs
    const int mg = job / 48;                  // 0..511 (64-row group)
    const int ng = job % 48;                  // 0..47  (32-col group)
    const int m0 = mg * 64;
    const int n0 = ng * 32;
    const int r0 = lane & 15;

    v8f acc[8];
#pragma unroll
    for (int j = 0; j < 8; ++j) acc[j] = zero8();

    v16bf aA[4], bA[2], aB[4], bB[2];
#pragma unroll
    for (int mi = 0; mi < 4; ++mi) aA[mi] = load_frag_a(A, I_DIM, m0 + mi * 16 + r0, 0, lane);
#pragma unroll
    for (int ni = 0; ni < 2; ++ni) bA[ni] = load_frag_b(W, I_DIM, n0 + ni * 16 + r0, 0, lane);

    for (int kk = 0; kk < (I_DIM / 64) - 1; ++kk) {   // 7 iterations
        const int k0 = kk * 64;
        // stage B loads (k0+32)
#pragma unroll
        for (int mi = 0; mi < 4; ++mi)
            aB[mi] = load_frag_a(A, I_DIM, m0 + mi * 16 + r0, k0 + 32, lane);
#pragma unroll
        for (int ni = 0; ni < 2; ++ni)
            bB[ni] = load_frag_b(W, I_DIM, n0 + ni * 16 + r0, k0 + 32, lane);
        // consume stage A (k0)
#pragma unroll
        for (int mi = 0; mi < 4; ++mi)
#pragma unroll
            for (int ni = 0; ni < 2; ++ni)
                acc[mi * 2 + ni] = wmma_bf16(aA[mi], bA[ni], acc[mi * 2 + ni]);
        // stage A loads (k0+64)
#pragma unroll
        for (int mi = 0; mi < 4; ++mi)
            aA[mi] = load_frag_a(A, I_DIM, m0 + mi * 16 + r0, k0 + 64, lane);
#pragma unroll
        for (int ni = 0; ni < 2; ++ni)
            bA[ni] = load_frag_b(W, I_DIM, n0 + ni * 16 + r0, k0 + 64, lane);
        // consume stage B (k0+32)
#pragma unroll
        for (int mi = 0; mi < 4; ++mi)
#pragma unroll
            for (int ni = 0; ni < 2; ++ni)
                acc[mi * 2 + ni] = wmma_bf16(aB[mi], bB[ni], acc[mi * 2 + ni]);
    }
    // tail: stage A holds k = I_DIM-64, load stage B at k = I_DIM-32
#pragma unroll
    for (int mi = 0; mi < 4; ++mi)
        aB[mi] = load_frag_a(A, I_DIM, m0 + mi * 16 + r0, I_DIM - 32, lane);
#pragma unroll
    for (int ni = 0; ni < 2; ++ni)
        bB[ni] = load_frag_b(W, I_DIM, n0 + ni * 16 + r0, I_DIM - 32, lane);
#pragma unroll
    for (int mi = 0; mi < 4; ++mi)
#pragma unroll
        for (int ni = 0; ni < 2; ++ni)
            acc[mi * 2 + ni] = wmma_bf16(aA[mi], bA[ni], acc[mi * 2 + ni]);
#pragma unroll
    for (int mi = 0; mi < 4; ++mi)
#pragma unroll
        for (int ni = 0; ni < 2; ++ni)
            acc[mi * 2 + ni] = wmma_bf16(aB[mi], bB[ni], acc[mi * 2 + ni]);

#pragma unroll
    for (int mi = 0; mi < 4; ++mi) {
        const int mbase = m0 + mi * 16 + ((lane < 16) ? 0 : 8);
#pragma unroll
        for (int ni = 0; ni < 2; ++ni) {
            const int n = n0 + ni * 16 + r0;
            const float bv = bias[n];
#pragma unroll
            for (int r = 0; r < 8; ++r) {
                gx[(size_t)(mbase + r) * G3 + n] = acc[mi * 2 + ni][r] + bv;
            }
        }
    }
}

__device__ __forceinline__ void grid_barrier(unsigned* __restrict__ cnt, unsigned target) {
    __builtin_amdgcn_fence(__ATOMIC_RELEASE, "agent");
    __syncthreads();
    if (threadIdx.x == 0) {
        __hip_atomic_fetch_add(cnt, 1u, __ATOMIC_RELAXED, __HIP_MEMORY_SCOPE_AGENT);
        while (__hip_atomic_load(cnt, __ATOMIC_RELAXED, __HIP_MEMORY_SCOPE_AGENT) < target) {
            __builtin_amdgcn_s_sleep(1);
        }
    }
    __syncthreads();
    __builtin_amdgcn_fence(__ATOMIC_ACQUIRE, "agent");
}

// Persistent scan kernel, NWG workgroups x 256 threads, one grid barrier per
// step. Each step: TDM the full 64KB bf16 h into LDS (wave 0 issues
// tensor_load_to_lds, waits tensorcnt), then a 64x96x512 bf16 WMMA GEMM with
// A-fragments from LDS and ping-pong pipelined W_hh fragments from L2.
// Dynamic LDS layout: h_lds 64KB | gh_lds 96*BSTR f32 | h_state 64*HC f32.
__global__ void __launch_bounds__(256) scan_kernel(
    const float* __restrict__ gx, const float* __restrict__ mask,
    const float* __restrict__ hidden, const __bf16* __restrict__ Whh,
    const float* __restrict__ b_hh,
    __bf16* __restrict__ hbuf0, __bf16* __restrict__ hbuf1,
    float* __restrict__ outputs, float* __restrict__ h_last,
    unsigned* __restrict__ barrier_cnt) {
    extern __shared__ char smem[];
    __bf16* h_lds  = (__bf16*)smem;                                  // 65536 B
    float*  gh_lds = (float*)(smem + 65536);                         // 96*BSTR*4 B
    float*  h_state = (float*)(smem + 65536 + 3 * HC * BSTR * 4);    // 64*HC*4 B

    const int lane = threadIdx.x & 31;
    const int wave = threadIdx.x >> 5;
    const int cb   = blockIdx.x * HC;

    // elementwise mapping: 64 batch x HC cols, 8 per thread
    const int eb  = threadIdx.x >> 2;
    const int ej0 = (threadIdx.x & 3) * 8;

    // init: own chunk of h into LDS state (f32) and into hbuf1 (bf16, read at t=0)
#pragma unroll
    for (int q = 0; q < 8; ++q) {
        const float hv = hidden[(size_t)eb * H_DIM + cb + ej0 + q];
        h_state[eb * HC + ej0 + q] = hv;
        hbuf1[(size_t)eb * H_DIM + cb + ej0 + q] = (__bf16)hv;
    }

    // GEMM mapping: 4 m-tiles x 6 n-tiles over 8 waves (3 n-tiles each)
    const int mt   = wave >> 1;
    const int nt0  = (wave & 1) * 3;
    const int r0   = lane & 15;
    const int arow = mt * 16 + r0;

    int grow[3];
#pragma unroll
    for (int j = 0; j < 3; ++j) {
        const int gl = (nt0 + j) * 16 + r0;      // 0..3*HC-1
        grow[j] = (gl / HC) * H_DIM + cb + (gl % HC);
    }

    for (int t = 0; t < T_DIM; ++t) {
        grid_barrier(barrier_cnt, (unsigned)(t + 1) * NWG);

        const __bf16* hread  = (t & 1) ? hbuf0 : hbuf1;
        __bf16*       hwrite = (t & 1) ? hbuf1 : hbuf0;

        // DMA previous h (64x512 bf16 = 64KB, contiguous) into LDS
        if (wave == 0) {
            tdm_load_1d(h_lds, hread, (unsigned)(B_DIM * H_DIM));
            __builtin_amdgcn_s_wait_tensorcnt(0);
        }
        __syncthreads();

        // gh slice: M=64 (batch), N=3*HC (this WG's gate rows), K=H
        v8f acc[3];
#pragma unroll
        for (int j = 0; j < 3; ++j) acc[j] = zero8();

        v16bf aA, bA[3], aB, bB[3];
        aA = load_frag_a_lds(h_lds, H_DIM, arow, 0, lane);
#pragma unroll
        for (int j = 0; j < 3; ++j) bA[j] = load_frag_b(Whh, H_DIM, grow[j], 0, lane);

        for (int kk = 0; kk < (H_DIM / 64) - 1; ++kk) {   // 7 iterations
            const int k0 = kk * 64;
            aB = load_frag_a_lds(h_lds, H_DIM, arow, k0 + 32, lane);
#pragma unroll
            for (int j = 0; j < 3; ++j) bB[j] = load_frag_b(Whh, H_DIM, grow[j], k0 + 32, lane);
#pragma unroll
            for (int j = 0; j < 3; ++j) acc[j] = wmma_bf16(aA, bA[j], acc[j]);
            aA = load_frag_a_lds(h_lds, H_DIM, arow, k0 + 64, lane);
#pragma unroll
            for (int j = 0; j < 3; ++j) bA[j] = load_frag_b(Whh, H_DIM, grow[j], k0 + 64, lane);
#pragma unroll
            for (int j = 0; j < 3; ++j) acc[j] = wmma_bf16(aB, bB[j], acc[j]);
        }
        aB = load_frag_a_lds(h_lds, H_DIM, arow, H_DIM - 32, lane);
#pragma unroll
        for (int j = 0; j < 3; ++j) bB[j] = load_frag_b(Whh, H_DIM, grow[j], H_DIM - 32, lane);
#pragma unroll
        for (int j = 0; j < 3; ++j) acc[j] = wmma_bf16(aA, bA[j], acc[j]);
#pragma unroll
        for (int j = 0; j < 3; ++j) acc[j] = wmma_bf16(aB, bB[j], acc[j]);

        // D fragment -> LDS: lane holds gate-row column gl, batch rows mb..mb+7
        const int mb = mt * 16 + ((lane < 16) ? 0 : 8);
#pragma unroll
        for (int j = 0; j < 3; ++j) {
            const int gl = (nt0 + j) * 16 + r0;
#pragma unroll
            for (int r = 0; r < 8; ++r) {
                gh_lds[gl * BSTR + mb + r] = acc[j][r];
            }
        }
        __syncthreads();

        // gate math (f32)
        const float  m      = mask[(size_t)t * B_DIM + eb];
        const size_t gxbase = ((size_t)t * B_DIM + eb) * G3 + cb + ej0;
        const size_t obase  = ((size_t)t * B_DIM + eb) * H_DIM + cb + ej0;
#pragma unroll
        for (int q = 0; q < 8; ++q) {
            const int j = ej0 + q;
            const float ghr = gh_lds[j * BSTR + eb]            + b_hh[cb + j];
            const float ghz = gh_lds[(HC + j) * BSTR + eb]     + b_hh[H_DIM + cb + j];
            const float ghn = gh_lds[(2 * HC + j) * BSTR + eb] + b_hh[2 * H_DIM + cb + j];
            const float gxr = gx[gxbase + q];
            const float gxz = gx[gxbase + H_DIM + q];
            const float gxn = gx[gxbase + 2 * H_DIM + q];
            const float r  = 1.f / (1.f + __expf(-(gxr + ghr)));
            const float z  = 1.f / (1.f + __expf(-(gxz + ghz)));
            const float nn = tanhf(gxn + r * ghn);
            const float hp = h_state[eb * HC + j];
            const float ht = (1.f - z) * nn + z * hp;
            const float hn = ht * m + hp * (1.f - m);
            outputs[obase + q] = ht * m;
            h_state[eb * HC + j] = hn;
            hwrite[(size_t)eb * H_DIM + cb + j] = (__bf16)hn;
        }
        // next iteration's grid_barrier (internal __syncthreads) orders LDS
        // reuse and the ping-pong swap
    }

#pragma unroll
    for (int q = 0; q < 8; ++q) {
        h_last[(size_t)eb * H_DIM + cb + ej0 + q] = h_state[eb * HC + ej0 + q];
    }
}

extern "C" void kernel_launch(void* const* d_in, const int* in_sizes, int n_in,
                              void* d_out, int out_size, void* d_ws, size_t ws_size,
                              hipStream_t stream) {
    (void)in_sizes; (void)n_in; (void)out_size; (void)ws_size;
    const float* seq_input = (const float*)d_in[0];
    const float* seq_mask  = (const float*)d_in[1];
    const float* hidden    = (const float*)d_in[2];
    const float* W_ih      = (const float*)d_in[3];
    const float* W_hh      = (const float*)d_in[4];
    const float* b_ih      = (const float*)d_in[5];
    const float* b_hh      = (const float*)d_in[6];

    float* out    = (float*)d_out;
    float* h_last = out + (size_t)T_DIM * B_DIM * H_DIM;

    char* ws = (char*)d_ws;
    __bf16* A_bf16 = (__bf16*)ws;  ws += (size_t)T_DIM * B_DIM * I_DIM * 2;
    __bf16* Wih_bf = (__bf16*)ws;  ws += (size_t)G3 * I_DIM * 2;
    __bf16* Whh_bf = (__bf16*)ws;  ws += (size_t)G3 * H_DIM * 2;
    __bf16* hbuf0  = (__bf16*)ws;  ws += (size_t)B_DIM * H_DIM * 2;
    __bf16* hbuf1  = (__bf16*)ws;  ws += (size_t)B_DIM * H_DIM * 2;
    unsigned* cnt  = (unsigned*)ws; ws += 256;
    float* gx      = (float*)ws;   // T*B*G3 f32 = 192 MB

    init_counter_kernel<<<1, 64, 0, stream>>>(cnt);

    const int n_seq = T_DIM * B_DIM * I_DIM;
    convert_bf16_kernel<<<(n_seq / 4 + 255) / 256, 256, 0, stream>>>(seq_input, A_bf16, n_seq);
    const int n_w = G3 * I_DIM;
    convert_bf16_kernel<<<(n_w / 4 + 255) / 256, 256, 0, stream>>>(W_ih, Wih_bf, n_w);
    convert_bf16_kernel<<<(n_w / 4 + 255) / 256, 256, 0, stream>>>(W_hh, Whh_bf, n_w);

    // 512 m-groups * 48 n-groups = 24576 wave jobs / 8 waves per block
    gx_gemm_kernel<<<3072, 256, 0, stream>>>(A_bf16, Wih_bf, b_ih, gx);

    const size_t smem_bytes = 65536 + (size_t)(3 * HC) * BSTR * 4 + (size_t)B_DIM * HC * 4;
    scan_kernel<<<NWG, 256, smem_bytes, stream>>>(gx, seq_mask, hidden, Whh_bf, b_hh,
                                                  hbuf0, hbuf1, out, h_last, cnt);
}
